// ByteMultiHeadSelfAttention_21131239097176
// MI455X (gfx1250) — matrix operand order repaired and checked by reading, not compiled, not run on hardware
//
#include <hip/hip_runtime.h>
#include <hip/hip_bf16.h>

#define B_ 2
#define T_ 2048
#define E_ 2048
#define H_ 16
#define KV_ 4
#define D_ 128
#define WINDOW_ 512
#define TPAD_ 2112   // T + padding so last 32-key chunk over-reads stay in-buffer

typedef __attribute__((ext_vector_type(16))) __bf16 v16bf;
typedef __attribute__((ext_vector_type(8)))  float  v8f;
typedef int vint4 __attribute__((vector_size(16)));  // payload type for async b128

#define AS1 __attribute__((address_space(1)))
#define AS3 __attribute__((address_space(3)))

#if defined(__has_builtin)
#if __has_builtin(__builtin_amdgcn_global_load_async_to_lds_b128) && \
    __has_builtin(__builtin_amdgcn_s_wait_asynccnt)
#define HAVE_ASYNC_LDS 1
#endif
#endif
#ifndef HAVE_ASYNC_LDS
#define HAVE_ASYNC_LDS 0
#endif

// ---- WMMA fragment loaders (CDNA5 16-bit layouts, wave32) -------------------
// A matrix 16x32 (MxK): lanes 0-15 row M=lane, elems 0..7 = K kb..kb+7,
// elems 8..15 = K 16+kb..16+kb+7 with kb = 0 (lanes<16) or 8 (lanes>=16).
__device__ __forceinline__ v16bf load_a_frag(const __bf16* p, int ld) {
    int lane = threadIdx.x & 31;
    int row  = lane & 15;
    int kb   = (lane >> 4) << 3;
    v16bf a;
    reinterpret_cast<uint4*>(&a)[0] = *reinterpret_cast<const uint4*>(p + row * ld + kb);
    reinterpret_cast<uint4*>(&a)[1] = *reinterpret_cast<const uint4*>(p + row * ld + 16 + kb);
    return a;
}

// B matrix 32x16 (KxN) loaded from a ROW-MAJOR [N x K] operand (i.e. B = opT):
// lane n = lane&15, elems 0..7 = K kb..kb+7, elems 8..15 = K 16+kb..16+kb+7 (kb = 0 or 8).
__device__ __forceinline__ v16bf load_bt_frag(const __bf16* p, int ld) {
    int lane = threadIdx.x & 31;
    int n    = lane & 15;
    int kb   = (lane >> 4) << 3;
    const __bf16* r = p + (size_t)n * ld + kb;
    v16bf b;
    reinterpret_cast<uint4*>(&b)[0] = *reinterpret_cast<const uint4*>(r);
    reinterpret_cast<uint4*>(&b)[1] = *reinterpret_cast<const uint4*>(r + 16);
    return b;
}

// ---- f32 -> bf16 conversion -------------------------------------------------
__global__ void cvt_f32_bf16(const float* __restrict__ in, __bf16* __restrict__ out, int n) {
    int i = blockIdx.x * blockDim.x + threadIdx.x;
    if (i < n) out[i] = (__bf16)in[i];
}

// ---- Generic GEMM: C[M,N] = A[M,K] * W[N,K]^T, bf16 in, f32 accum ----------
template <bool OUT_F32>
__global__ __launch_bounds__(256, 1) void gemm_xwt(const __bf16* __restrict__ A,
                                                   const __bf16* __restrict__ W,
                                                   void* __restrict__ out,
                                                   int M, int N, int K) {
    int wave = threadIdx.x >> 5;
    int lane = threadIdx.x & 31;
    int m0 = blockIdx.y * 64 + (wave & 3) * 16;
    int n0 = blockIdx.x * 32 + (wave >> 2) * 16;
    const __bf16* ap = A + (size_t)m0 * K;
    const __bf16* wp = W + (size_t)n0 * K;
    v8f acc = {};
#pragma unroll 4
    for (int kk = 0; kk < K; kk += 32) {
        __builtin_prefetch(ap + kk + 512, 0, 1);   // global_prefetch_b8
        __builtin_prefetch(wp + kk + 512, 0, 1);
        v16bf a = load_a_frag(ap + kk, K);
        v16bf b = load_bt_frag(wp + kk, K);
        acc = __builtin_amdgcn_wmma_f32_16x16x32_bf16(false, a, false, b,
                                                      (short)0, acc, false, false);
    }
    int n  = lane & 15;
    int mh = (lane >> 4) << 3;
    if (OUT_F32) {
        float* o = (float*)out;
#pragma unroll
        for (int r = 0; r < 8; ++r)
            o[(size_t)(m0 + mh + r) * N + n0 + n] = acc[r];
    } else {
        __bf16* o = (__bf16*)out;
#pragma unroll
        for (int r = 0; r < 8; ++r)
            o[(size_t)(m0 + mh + r) * N + n0 + n] = (__bf16)acc[r];
    }
}

// ---- RoPE + relayout: in [B,T,NH,128] -> out [B,NH,tpad,128] ---------------
__global__ void rope_kernel(const __bf16* __restrict__ in, __bf16* __restrict__ out,
                            int NH, int tpad_out, int total) {
    int idx = blockIdx.x * blockDim.x + threadIdx.x;
    if (idx >= total) return;
    int d = idx % 64;
    int h = (idx / 64) % NH;
    int t = (idx / (64 * NH)) % T_;
    int b = idx / (64 * NH * T_);
    float inv = __powf(10000.0f, -(float)d / 64.0f);
    float ang = (float)t * inv;
    float c = __cosf(ang), s = __sinf(ang);
    size_t ib = ((size_t)(b * T_ + t) * NH + h) * D_;
    float x1 = (float)in[ib + d];
    float x2 = (float)in[ib + d + 64];
    size_t ob = ((size_t)(b * NH + h) * tpad_out + t) * D_;
    out[ob + d]      = (__bf16)(x1 * c - x2 * s);
    out[ob + d + 64] = (__bf16)(x2 * c + x1 * s);
}

// ---- V transpose: [B,T,KV,128] -> [B,KV,128,TPAD] ---------------------------
__global__ void vtrans_kernel(const __bf16* __restrict__ in, __bf16* __restrict__ out, int total) {
    int idx = blockIdx.x * blockDim.x + threadIdx.x;
    if (idx >= total) return;
    int d   = idx % D_;
    int kvh = (idx / D_) % KV_;
    int t   = (idx / (D_ * KV_)) % T_;
    int b   = idx / (D_ * KV_ * T_);
    out[((size_t)(b * KV_ + kvh) * D_ + d) * TPAD_ + t] =
        in[((size_t)(b * T_ + t) * KV_ + kvh) * D_ + d];
}

#if HAVE_ASYNC_LDS
// Cooperatively stage a 32x128 bf16 K chunk (8 KB) into LDS with the CDNA5
// async DMA path: 128 threads x 64 B each, four async b128 ops per thread.
__device__ __forceinline__ void stage_k_async(const __bf16* gsrc, __bf16* ldst) {
    int byteoff = threadIdx.x * 64;
    const AS1 char* g = (const AS1 char*)gsrc + byteoff;
    AS3 char*       l = (AS3 char*)ldst + byteoff;
#pragma unroll
    for (int i = 0; i < 4; ++i)
        __builtin_amdgcn_global_load_async_to_lds_b128(
            (AS1 vint4*)(g + 16 * i), (AS3 vint4*)(l + 16 * i), 0, 0);
}
#endif

// ---- Flash attention, sliding-window causal, GQA ---------------------------
// q: [B,H,T,128], k: [B,KV,TPAD,128] (rope'd), vT: [B,KV,128,TPAD]
// ao: [B,T,H*128] bf16.  1 wave per 16-row query tile; 4 waves/block share the
// same KV head AND the same t0 => identical chunk schedule => block-uniform
// barriers are legal and K chunks can be staged into LDS once per block.
__global__ __launch_bounds__(128, 1) void flash_attn_kernel(const __bf16* __restrict__ q,
                                                            const __bf16* __restrict__ k,
                                                            const __bf16* __restrict__ vT,
                                                            __bf16* __restrict__ ao) {
    __shared__ __bf16 lds_q[4][16 * 128];  // per-wave Q tile (written once)
    __shared__ __bf16 lds_p[4][16 * 32];   // per-wave exp(S) staging for P·V
#if HAVE_ASYNC_LDS
    __shared__ __bf16 lds_k[2][32 * 128];  // block-shared double-buffered K chunk
#endif
    int wave = threadIdx.x >> 5;
    int lane = threadIdx.x & 31;
    int hg = blockIdx.x % (H_ / 4);
    int qt = (blockIdx.x / (H_ / 4)) % (T_ / 16);
    int b  = blockIdx.x / ((H_ / 4) * (T_ / 16));
    int h  = hg * 4 + wave;
    int kvh = h >> 2;
    int t0 = qt * 16;

    const __bf16* qp = q  + ((size_t)(b * H_ + h) * T_ + t0) * D_;
    const __bf16* kp = k  + (size_t)(b * KV_ + kvh) * TPAD_ * D_;
    const __bf16* vp = vT + (size_t)(b * KV_ + kvh) * D_ * TPAD_;

    __bf16* qbuf = lds_q[wave];
    __bf16* pbuf = lds_p[wave];

    // Stage the 16x128 Q tile into LDS (wave-private; in-wave DS ordering).
    {
        int row = lane >> 1;
        int c0  = (lane & 1) * 64;
        const uint4* src = reinterpret_cast<const uint4*>(qp + row * D_ + c0);
        uint4* dst = reinterpret_cast<uint4*>(qbuf + row * 128 + c0);
        dst[0] = src[0]; dst[1] = src[1]; dst[2] = src[2]; dst[3] = src[3];
    }

    v8f o[8] = {};
    float mrow[8], lrow[8];
#pragma unroll
    for (int r = 0; r < 8; ++r) { mrow[r] = -3.0e38f; lrow[r] = 0.0f; }

    const float scale = 0.088388347648318447f;  // 1/sqrt(128)
    int wstart = t0 - WINDOW_ + 1;
    int kstart = wstart > 0 ? (wstart & ~31) : 0;
    int kend   = t0 + 15;
    int n  = lane & 15;
    int mh = (lane >> 4) << 3;

#if HAVE_ASYNC_LDS
    stage_k_async(kp + (size_t)kstart * D_, lds_k[0]);
    __builtin_amdgcn_s_wait_asynccnt(0);
    __syncthreads();
    int pb = 0;
#endif

    for (int kc = kstart; kc <= kend; kc += 32) {
#if HAVE_ASYNC_LDS
        const __bf16* kc0 = lds_k[pb];             // keys kc..kc+15
        const __bf16* kc1 = lds_k[pb] + 16 * 128;  // keys kc+16..kc+31
        if (kc + 32 <= kend)                       // DMA next chunk during compute
            stage_k_async(kp + (size_t)(kc + 32) * D_, lds_k[pb ^ 1]);
#else
        const __bf16* kc0 = kp + (size_t)kc * D_;
        const __bf16* kc1 = kp + (size_t)(kc + 16) * D_;
#endif
        // Load all eight K^T fragments, then run the 8 QK^T WMMAs.
        v16bf bk[8];
#pragma unroll
        for (int j = 0; j < 4; ++j) {
            bk[j]     = load_bt_frag(kc0 + 32 * j, 128);
            bk[4 + j] = load_bt_frag(kc1 + 32 * j, 128);
        }
        v8f s0 = {}, s1 = {};
#pragma unroll
        for (int j = 0; j < 4; ++j) {
            v16bf aj = load_a_frag(qbuf + 32 * j, 128);
            s0 = __builtin_amdgcn_wmma_f32_16x16x32_bf16(false, aj, false, bk[j],
                                                         (short)0, s0, false, false);
            s1 = __builtin_amdgcn_wmma_f32_16x16x32_bf16(false, aj, false, bk[4 + j],
                                                         (short)0, s1, false, false);
        }
        float sf[8];
#pragma unroll
        for (int r = 0; r < 8; ++r) {
            int t = t0 + mh + r;
            int key0 = kc + n, key1 = kc + 16 + n;
            bool v0 = (key0 <= t) && (t - key0) < WINDOW_;
            bool v1 = (key1 <= t) && (t - key1) < WINDOW_;
            float x0 = v0 ? s0[r] * scale : -1.0e9f;
            float x1 = v1 ? s1[r] * scale : -1.0e9f;
            float cm = fmaxf(x0, x1);
            cm = fmaxf(cm, __shfl_xor(cm, 1));
            cm = fmaxf(cm, __shfl_xor(cm, 2));
            cm = fmaxf(cm, __shfl_xor(cm, 4));
            cm = fmaxf(cm, __shfl_xor(cm, 8));
            float mn = fmaxf(mrow[r], cm);
            float s  = __expf(mrow[r] - mn);
            float p0 = v0 ? __expf(x0 - mn) : 0.0f;
            float p1 = v1 ? __expf(x1 - mn) : 0.0f;
            float rs = p0 + p1;
            rs += __shfl_xor(rs, 1);
            rs += __shfl_xor(rs, 2);
            rs += __shfl_xor(rs, 4);
            rs += __shfl_xor(rs, 8);
            lrow[r] = lrow[r] * s + rs;
            mrow[r] = mn;
            sf[r] = s;
            pbuf[(mh + r) * 32 + n]      = (__bf16)p0;
            pbuf[(mh + r) * 32 + 16 + n] = (__bf16)p1;
        }
#pragma unroll
        for (int nt = 0; nt < 8; ++nt)
#pragma unroll
            for (int r = 0; r < 8; ++r) o[nt][r] *= sf[r];
        v16bf ap = load_a_frag(pbuf, 32);   // in-wave DS ordering covers store->load
#pragma unroll
        for (int nt = 0; nt < 8; ++nt) {
            v16bf bv = load_bt_frag(vp + (size_t)(nt * 16) * TPAD_ + kc, TPAD_);
            o[nt] = __builtin_amdgcn_wmma_f32_16x16x32_bf16(false, ap, false, bv,
                                                            (short)0, o[nt], false, false);
        }
#if HAVE_ASYNC_LDS
        if (kc + 32 <= kend) __builtin_amdgcn_s_wait_asynccnt(0);
        __syncthreads();   // uniform: all waves share kstart/kend
        pb ^= 1;
#endif
    }

#pragma unroll
    for (int nt = 0; nt < 8; ++nt)
#pragma unroll
        for (int r = 0; r < 8; ++r) {
            int t = t0 + mh + r;
            int col = h * D_ + nt * 16 + n;
            ao[(size_t)(b * T_ + t) * E_ + col] = (__bf16)(o[nt][r] / lrow[r]);
        }
}

extern "C" void kernel_launch(void* const* d_in, const int* in_sizes, int n_in,
                              void* d_out, int out_size, void* d_ws, size_t ws_size,
                              hipStream_t stream) {
    const float* x  = (const float*)d_in[0];
    const float* Wq = (const float*)d_in[1];
    const float* Wk = (const float*)d_in[2];
    const float* Wv = (const float*)d_in[3];
    const float* Wo = (const float*)d_in[4];

    constexpr size_t NX  = (size_t)B_ * T_ * E_;        // 8388608
    constexpr size_t NWQ = (size_t)E_ * E_;             // 4194304
    constexpr size_t NWK = (size_t)(KV_ * D_) * E_;     // 1048576
    constexpr size_t NQ  = (size_t)B_ * T_ * H_ * D_;   // 8388608
    constexpr size_t NK  = (size_t)B_ * T_ * KV_ * D_;  // 2097152
    constexpr size_t NKA = (size_t)B_ * KV_ * TPAD_ * D_;

    char* p = (char*)d_ws;
    size_t off = 0;
    auto take = [&](size_t elems) {
        __bf16* r = (__bf16*)(p + off);
        off += ((elems * sizeof(__bf16)) + 255) & ~(size_t)255;
        return r;
    };
    __bf16* xb  = take(NX);
    __bf16* wqb = take(NWQ);
    __bf16* wkb = take(NWK);
    __bf16* wvb = take(NWK);
    __bf16* wob = take(NWQ);
    __bf16* qb  = take(NQ);    // pre-rope projections
    __bf16* kb  = take(NK);
    __bf16* vb  = take(NK);
    __bf16* qa  = take(NQ);    // [B,H,T,D]
    __bf16* ka  = take(NKA);   // [B,KV,TPAD,D]
    __bf16* vt  = take(NKA);   // [B,KV,D,TPAD]
    __bf16* ao  = take(NX);    // [B,T,E]

    const int CT = 256;
    cvt_f32_bf16<<<(NX  + CT - 1) / CT, CT, 0, stream>>>(x,  xb,  (int)NX);
    cvt_f32_bf16<<<(NWQ + CT - 1) / CT, CT, 0, stream>>>(Wq, wqb, (int)NWQ);
    cvt_f32_bf16<<<(NWK + CT - 1) / CT, CT, 0, stream>>>(Wk, wkb, (int)NWK);
    cvt_f32_bf16<<<(NWK + CT - 1) / CT, CT, 0, stream>>>(Wv, wvb, (int)NWK);
    cvt_f32_bf16<<<(NWQ + CT - 1) / CT, CT, 0, stream>>>(Wo, wob, (int)NWQ);

    const int M = B_ * T_;  // 4096
    gemm_xwt<false><<<dim3(E_ / 32, M / 64), 256, 0, stream>>>(xb, wqb, qb, M, E_, E_);
    gemm_xwt<false><<<dim3((KV_ * D_) / 32, M / 64), 256, 0, stream>>>(xb, wkb, kb, M, KV_ * D_, E_);
    gemm_xwt<false><<<dim3((KV_ * D_) / 32, M / 64), 256, 0, stream>>>(xb, wvb, vb, M, KV_ * D_, E_);

    int rq = B_ * T_ * H_ * 64;
    int rk = B_ * T_ * KV_ * 64;
    int tv = B_ * T_ * KV_ * D_;
    rope_kernel<<<(rq + CT - 1) / CT, CT, 0, stream>>>(qb, qa, H_,  T_,    rq);
    rope_kernel<<<(rk + CT - 1) / CT, CT, 0, stream>>>(kb, ka, KV_, TPAD_, rk);
    vtrans_kernel<<<(tv + CT - 1) / CT, CT, 0, stream>>>(vb, vt, tv);

    flash_attn_kernel<<<B_ * (T_ / 16) * (H_ / 4), 128, 0, stream>>>(qa, ka, vt, ao);

    gemm_xwt<true><<<dim3(E_ / 32, M / 64), 256, 0, stream>>>(ao, wob, d_out, M, E_, E_);
}
